// MoELayer_23905787969930
// MI455X (gfx1250) — compile-verified, matrix operand order checked by reading, not run on hardware
//
#include <hip/hip_runtime.h>
#include <hip/hip_bf16.h>

#define NEXP 8
#define DIM  1024
#define FDIM 2048
#define NTOK 4096            // B*S
#define MT1  (NTOK / 128)    // 32 max M-tiles per expert (128-row tiles)

typedef __bf16 bf16;
typedef __attribute__((ext_vector_type(16))) __bf16 v16bf;
typedef __attribute__((ext_vector_type(8)))  float  v8f;

// Load one 16x32 bf16 WMMA fragment from an LDS tile row (stride 40 bf16).
// Per ISA layout: lane holds K[8g..8g+8) in v0-3 and K[16+8g..16+8g+8) in v4-7.
__device__ __forceinline__ v16bf load_frag16(const bf16* p) {
  union { uint4 u[2]; v16bf v; } u;
  u.u[0] = *(const uint4*)(p);
  u.u[1] = *(const uint4*)(p + 16);
  return u.v;
}

__device__ __forceinline__ v8f wmma_bf16(v16bf a, v16bf b, v8f c) {
  // (neg_a, A, neg_b, B, c_mod, C, reuse_a, reuse_b)
  return __builtin_amdgcn_wmma_f32_16x16x32_bf16(false, a, false, b, (short)0, c,
                                                 false, false);
}

__device__ __forceinline__ float fast_sigmoid(float x) {
#if __has_builtin(__builtin_amdgcn_rcpf)
  return __builtin_amdgcn_rcpf(1.f + __expf(-x));
#else
  return 1.f / (1.f + __expf(-x));
#endif
}

// gfx1250 async global->LDS copy (ASYNCcnt path). lds_off = wave LDS byte
// address (low 32 bits of generic pointer), gptr = 64-bit global address.
__device__ __forceinline__ void async_ld_b128(unsigned lds_off, const void* gptr) {
  asm volatile("global_load_async_to_lds_b128 %0, %1, off"
               :: "v"(lds_off), "v"(gptr) : "memory");
}
__device__ __forceinline__ void wait_async0() {
#if __has_builtin(__builtin_amdgcn_s_wait_asynccnt)
  __builtin_amdgcn_s_wait_asynccnt(0);
#else
  asm volatile("s_wait_asynccnt 0x0" ::: "memory");
#endif
}

// ---------------------------------------------------------------------------
// Kernel 1: router. One thread per token: gate GEMV, top-2, push to lists.
// ---------------------------------------------------------------------------
__global__ void moe_router(const float* __restrict__ x, const float* __restrict__ gw,
                           int* __restrict__ cnt, int* __restrict__ list,
                           int* __restrict__ pidx, float* __restrict__ wlist) {
  int t = blockIdx.x * blockDim.x + threadIdx.x;
  if (t >= NTOK) return;
  float acc[NEXP];
#pragma unroll
  for (int e = 0; e < NEXP; ++e) acc[e] = 0.f;
  const float* xr = x + (size_t)t * DIM;
  for (int d = 0; d < DIM; d += 4) {
    float4 xv = *(const float4*)(xr + d);
    float xs[4] = {xv.x, xv.y, xv.z, xv.w};
#pragma unroll
    for (int j = 0; j < 4; ++j) {
      const float* g = gw + (size_t)(d + j) * NEXP;
#pragma unroll
      for (int e = 0; e < NEXP; ++e) acc[e] += xs[j] * g[e];
    }
  }
  int i0 = 0;
#pragma unroll
  for (int e = 1; e < NEXP; ++e) if (acc[e] > acc[i0]) i0 = e;
  int i1 = (i0 == 0) ? 1 : 0;
#pragma unroll
  for (int e = 0; e < NEXP; ++e) if (e != i0 && acc[e] > acc[i1]) i1 = e;
  float e1 = __expf(acc[i1] - acc[i0]);     // softmax denominator cancels
  float w0 = 1.f / (1.f + e1);
  float w1 = 1.f - w0;
  int s0 = atomicAdd(&cnt[i0], 1);
  list[i0 * NTOK + s0] = t; pidx[i0 * NTOK + s0] = t * 2;     wlist[i0 * NTOK + s0] = w0;
  int s1 = atomicAdd(&cnt[i1], 1);
  list[i1 * NTOK + s1] = t; pidx[i1 * NTOK + s1] = t * 2 + 1; wlist[i1 * NTOK + s1] = w1;
}

// ---------------------------------------------------------------------------
// Kernel 2: act = silu(x@w1[e]) * (x@w3[e]) for routed rows (grouped GEMM).
// 512 threads = 16 waves. Block tile 128(M) x 128(N), K-step 32.
// Wave grid 4x4, wave tile 32x32 (2x2 fragments per matrix).
// ---------------------------------------------------------------------------
__global__ void __launch_bounds__(512, 1) moe_ffn1(
    const float* __restrict__ x, const float* __restrict__ w1,
    const float* __restrict__ w3, const int* __restrict__ cnt,
    const int* __restrict__ list, const int* __restrict__ pidx,
    bf16* __restrict__ act) {
  int e  = blockIdx.x / MT1;
  int mt = blockIdx.x % MT1;
  int rows = cnt[e];
  int m0 = mt * 128;
  if (m0 >= rows) return;
  int n0 = blockIdx.y * 128;

  // LDS pool: lA [128][40]; lB1/lB3 [128][40] each; epilogue stage [128][136]
  // overlays lB1+lB3.
  __shared__ alignas(16) char smem[46080];
  bf16 (*lA)[40]   = (bf16(*)[40])(smem);
  bf16 (*lB1)[40]  = (bf16(*)[40])(smem + 10240);
  bf16 (*lB3)[40]  = (bf16(*)[40])(smem + 20480);
  bf16 (*lS)[136]  = (bf16(*)[136])(smem + 10240);
  int* ltok = (int*)(smem + 45056);
  int* lpr  = (int*)(smem + 45056 + 512);

  int tid = threadIdx.x;
  if (tid < 128) {
    int s = m0 + tid; if (s > rows - 1) s = rows - 1;   // clamp padding rows
    ltok[tid] = list[e * NTOK + s];
    lpr[tid]  = pidx[e * NTOK + s];
  }
  __syncthreads();

  int lane = tid & 31;
  int wave = tid >> 5;
  int wm = (wave >> 2) * 32;   // wave M offset in block tile
  int wn = (wave & 3) * 32;    // wave N offset in block tile

  v8f acc1[2][2], acc3[2][2];
  v8f vz = {0.f, 0.f, 0.f, 0.f, 0.f, 0.f, 0.f, 0.f};
#pragma unroll
  for (int i = 0; i < 2; ++i)
#pragma unroll
    for (int j = 0; j < 2; ++j) { acc1[i][j] = vz; acc3[i][j] = vz; }

  const float* W1 = w1 + (size_t)e * DIM * FDIM;
  const float* W3 = w3 + (size_t)e * DIM * FDIM;

  int ar = tid >> 2;            // A-stage row 0..127
  int ac = (tid & 3) * 8;       // A-stage k offset (8 floats)
  const float* xrow = x + (size_t)ltok[ar] * DIM + ac;
  int bn = tid & 127;           // B-stage column 0..127
  int bk = (tid >> 7) * 8;      // B-stage k base (8 rows of K)

  for (int k0 = 0; k0 < DIM; k0 += 32) {
    // stage A tile (f32 -> bf16)
    {
      float4 v0 = *(const float4*)(xrow + k0);
      float4 v1 = *(const float4*)(xrow + k0 + 4);
      lA[ar][ac + 0] = (bf16)v0.x; lA[ar][ac + 1] = (bf16)v0.y;
      lA[ar][ac + 2] = (bf16)v0.z; lA[ar][ac + 3] = (bf16)v0.w;
      lA[ar][ac + 4] = (bf16)v1.x; lA[ar][ac + 5] = (bf16)v1.y;
      lA[ar][ac + 6] = (bf16)v1.z; lA[ar][ac + 7] = (bf16)v1.w;
    }
    // stage B tiles transposed (f32 -> bf16); thread owns one column, 8 k-rows
    {
      const float* p1 = W1 + (size_t)(k0 + bk) * FDIM + n0 + bn;
      const float* p3 = W3 + (size_t)(k0 + bk) * FDIM + n0 + bn;
      __builtin_prefetch(p1 + (size_t)32 * FDIM, 0, 0);
      __builtin_prefetch(p3 + (size_t)32 * FDIM, 0, 0);
#pragma unroll
      for (int j = 0; j < 8; ++j) {
        lB1[bn][bk + j] = (bf16)p1[(size_t)j * FDIM];
        lB3[bn][bk + j] = (bf16)p3[(size_t)j * FDIM];
      }
    }
    __syncthreads();

    int g8 = (lane >> 4) * 8;
    v16bf a0 = load_frag16(&lA[wm + (lane & 15)     ][g8]);
    v16bf a1 = load_frag16(&lA[wm + (lane & 15) + 16][g8]);
#pragma unroll
    for (int ni = 0; ni < 2; ++ni) {
      int col = wn + ni * 16 + (lane & 15);
      v16bf b1 = load_frag16(&lB1[col][g8]);
      v16bf b3 = load_frag16(&lB3[col][g8]);
      acc1[0][ni] = wmma_bf16(a0, b1, acc1[0][ni]);
      acc1[1][ni] = wmma_bf16(a1, b1, acc1[1][ni]);
      acc3[0][ni] = wmma_bf16(a0, b3, acc3[0][ni]);
      acc3[1][ni] = wmma_bf16(a1, b3, acc3[1][ni]);
    }
    __syncthreads();
  }

  // epilogue: silu(h1)*h3 -> LDS stage -> wide b128 global stores
#pragma unroll
  for (int mi = 0; mi < 2; ++mi) {
#pragma unroll
    for (int ni = 0; ni < 2; ++ni) {
      int c = wn + ni * 16 + (lane & 15);
#pragma unroll
      for (int v = 0; v < 8; ++v) {
        int r = wm + mi * 16 + (lane >> 4) * 8 + v;   // D-matrix row mapping
        float h1 = acc1[mi][ni][v];
        float h3 = acc3[mi][ni][v];
        lS[r][c] = (bf16)(h1 * fast_sigmoid(h1) * h3);
      }
    }
  }
  __syncthreads();
  {
    int r  = tid >> 2;
    int cb = (tid & 3) * 32;
    if (m0 + r < rows) {
      bf16* dst = act + (size_t)lpr[r] * FDIM + n0 + cb;
#pragma unroll
      for (int j = 0; j < 4; ++j)
        *(uint4*)(dst + j * 8) = *(const uint4*)(&lS[r][cb + j * 8]);
    }
  }
}

// ---------------------------------------------------------------------------
// Kernel 3: out += (act @ w2[e]) * route_weight (grouped GEMM + f32 atomics).
// 512 threads, block tile 128(M) x 128(N). A tile staged with async-to-LDS.
// ---------------------------------------------------------------------------
__global__ void __launch_bounds__(512, 1) moe_ffn2(
    const bf16* __restrict__ act, const float* __restrict__ w2,
    const int* __restrict__ cnt, const int* __restrict__ list,
    const int* __restrict__ pidx, const float* __restrict__ wlist,
    float* __restrict__ out) {
  int e  = blockIdx.x / MT1;
  int mt = blockIdx.x % MT1;
  int rows = cnt[e];
  int m0 = mt * 128;
  if (m0 >= rows) return;
  int n0 = blockIdx.y * 128;

  __shared__ alignas(16) bf16 lA[128][40];
  __shared__ alignas(16) bf16 lB[128][40];
  __shared__ int   ltok[128];
  __shared__ int   lpr[128];
  __shared__ float lw[128];

  int tid = threadIdx.x;
  if (tid < 128) {
    int s = m0 + tid; if (s > rows - 1) s = rows - 1;
    ltok[tid] = list[e * NTOK + s];
    lpr[tid]  = pidx[e * NTOK + s];
    lw[tid]   = wlist[e * NTOK + s];
  }
  __syncthreads();

  int lane = tid & 31;
  int wave = tid >> 5;
  int wm = (wave >> 2) * 32;
  int wn = (wave & 3) * 32;

  v8f acc[2][2];
  v8f vz = {0.f, 0.f, 0.f, 0.f, 0.f, 0.f, 0.f, 0.f};
#pragma unroll
  for (int i = 0; i < 2; ++i)
#pragma unroll
    for (int j = 0; j < 2; ++j) acc[i][j] = vz;

  const float* W2 = w2 + (size_t)e * FDIM * DIM;
  int ar = tid >> 2;                   // A row 0..127
  int ac = (tid & 3) * 8;              // A k offset (8 bf16 = 16B)
  const bf16* arow = act + (size_t)lpr[ar] * FDIM + ac;
  unsigned lA_off = (unsigned)(size_t)&lA[ar][ac];   // LDS byte address
  int bn = tid & 127;
  int bk = (tid >> 7) * 8;

  for (int k0 = 0; k0 < FDIM; k0 += 32) {
    // stage A tile via async global->LDS DMA (raw bf16 copy, ASYNCcnt)
    async_ld_b128(lA_off, arow + k0);
    // stage B tile transposed (f32 -> bf16) while the async copy is in flight
    {
      const float* p2 = W2 + (size_t)(k0 + bk) * DIM + n0 + bn;
      __builtin_prefetch(p2 + (size_t)32 * DIM, 0, 0);
#pragma unroll
      for (int j = 0; j < 8; ++j)
        lB[bn][bk + j] = (bf16)p2[(size_t)j * DIM];
    }
    wait_async0();
    __syncthreads();

    int g8 = (lane >> 4) * 8;
    v16bf a0 = load_frag16(&lA[wm + (lane & 15)     ][g8]);
    v16bf a1 = load_frag16(&lA[wm + (lane & 15) + 16][g8]);
#pragma unroll
    for (int ni = 0; ni < 2; ++ni) {
      int col = wn + ni * 16 + (lane & 15);
      v16bf b = load_frag16(&lB[col][g8]);
      acc[0][ni] = wmma_bf16(a0, b, acc[0][ni]);
      acc[1][ni] = wmma_bf16(a1, b, acc[1][ni]);
    }
    __syncthreads();
  }

#pragma unroll
  for (int mi = 0; mi < 2; ++mi) {
#pragma unroll
    for (int ni = 0; ni < 2; ++ni) {
      int col = n0 + wn + ni * 16 + (lane & 15);
#pragma unroll
      for (int v = 0; v < 8; ++v) {
        int r = wm + mi * 16 + (lane >> 4) * 8 + v;
        if (m0 + r < rows) {
          float val = acc[mi][ni][v] * lw[r];
          unsafeAtomicAdd(&out[(size_t)ltok[r] * DIM + col], val);
        }
      }
    }
  }
}

// ---------------------------------------------------------------------------
extern "C" void kernel_launch(void* const* d_in, const int* in_sizes, int n_in,
                              void* d_out, int out_size, void* d_ws, size_t ws_size,
                              hipStream_t stream) {
  const float* x  = (const float*)d_in[0];   // hidden_states [B,S,D]
  const float* gw = (const float*)d_in[1];   // gate_w [D,E]
  const float* w1 = (const float*)d_in[2];   // [E,D,F]
  const float* w2 = (const float*)d_in[3];   // [E,F,D]
  const float* w3 = (const float*)d_in[4];   // [E,D,F]
  float* out = (float*)d_out;

  char* ws = (char*)d_ws;
  const size_t LSZ = (size_t)4 * NEXP * NTOK;        // 128 KB per table
  int*   cnt   = (int*)(ws);
  int*   list  = (int*)(ws + 1024);
  int*   pidx  = (int*)(ws + 1024 + LSZ);
  float* wlist = (float*)(ws + 1024 + 2 * LSZ);
  bf16*  act   = (bf16*)(ws + 1024 + 3 * LSZ);       // [2*NTOK][FDIM] bf16 = 32 MB

  hipMemsetAsync(cnt, 0, NEXP * sizeof(int), stream);
  hipMemsetAsync(d_out, 0, (size_t)out_size * sizeof(float), stream);

  moe_router<<<NTOK / 256, 256, 0, stream>>>(x, gw, cnt, list, pidx, wlist);
  moe_ffn1<<<dim3(NEXP * MT1, FDIM / 128), 512, 0, stream>>>(x, w1, w3, cnt,
                                                             list, pidx, act);
  moe_ffn2<<<dim3(NEXP * MT1, DIM / 128), 512, 0, stream>>>(act, w2, cnt, list,
                                                            pidx, wlist, out);
}
